// MultibandFrameAttention_41214506173067
// MI455X (gfx1250) — matrix operand order repaired and checked by reading, not compiled, not run on hardware
//
#include <hip/hip_runtime.h>

// ---------------------------------------------------------------------------
// MultibandFrameAttention for MI455X (gfx1250, wave32, WMMA bf16)
//
//   b=4, w=1024, c=1024, h=16, d=64
//   q,k,v = x @ W^T + b            (bf16 WMMA GEMM, f32 accum)
//   S[i,j] = (q_i . k_j  +  erT_i . q_j) / sqrt(c)   <- bias is a 2nd WMMA
//   flash softmax over keys, O = P @ V, out = O @ Wo^T + bo (f32)
// ---------------------------------------------------------------------------

typedef __attribute__((ext_vector_type(16))) __bf16      v16bf;
typedef __attribute__((ext_vector_type(8)))  float       v8f;
typedef __attribute__((ext_vector_type(4)))  unsigned int v4u;

#define BINS    1024
#define CROP    1024
#define BATCH   4
#define HEADS   16
#define HEADDIM 64
#define ROWS    (BATCH * CROP)   // 4096 flattened (b,w) rows

#define WMMA_BF16(a, b, c) \
  __builtin_amdgcn_wmma_f32_16x16x32_bf16(false, (a), false, (b), (short)0, (c), false, false)

static __device__ __forceinline__ __bf16 to_bf16(float f) { return (__bf16)f; }

// A-operand frag, 16x32 bf16 tile of a row-major matrix (contiguous K).
// lanes 0-15: row = m0+lane,  K = k0+{0..7, 16..23}
// lanes16-31: row = m0+lane-16, K = k0+{8..15, 24..31}
static __device__ __forceinline__ v16bf
load_a_frag(const __bf16* __restrict__ base, int lda, int m0, int k0, int lane) {
  int hi = lane >> 4, mr = lane & 15;
  const __bf16* p = base + (size_t)(m0 + mr) * lda + k0 + (hi ? 8 : 0);
  union { v16bf bf; v4u u[2]; } u;
  u.u[0] = *(const v4u*)(p);
  u.u[1] = *(const v4u*)(p + 16);
  return u.bf;
}

// B-operand frag, 32x16 bf16 tile where B's column n is a contiguous row of
// `base` (stride ldb).  lanes 0-15: K=k0+{0..7,16..23}, lanes 16-31: K=k0+{8..15,24..31}.
static __device__ __forceinline__ v16bf
load_b_frag(const __bf16* __restrict__ base, int ldb, int k0, int n0, int lane) {
  int hi = lane >> 4, nr = lane & 15;
  const __bf16* p = base + (size_t)(n0 + nr) * ldb + k0 + (hi ? 8 : 0);
  union { v16bf bf; v4u u[2]; } u;
  u.u[0] = *(const v4u*)(p);
  u.u[1] = *(const v4u*)(p + 16);
  return u.bf;
}

// ---------------------------------------------------------------------------
// f32 -> bf16 conversion kernels
// ---------------------------------------------------------------------------
__global__ void cvt_bf16(const float* __restrict__ src, __bf16* __restrict__ dst, int n) {
  int i = blockIdx.x * blockDim.x + threadIdx.x;
  if (i < n) dst[i] = to_bf16(src[i]);
}

// er [64][1024] f32  ->  erT [1024][64] bf16  (rows contiguous in d)
__global__ void cvt_erT(const float* __restrict__ er, __bf16* __restrict__ erT) {
  int i = blockIdx.x * blockDim.x + threadIdx.x;
  if (i < HEADDIM * CROP) {
    int d = i >> 10, c = i & 1023;
    erT[c * HEADDIM + d] = to_bf16(er[i]);
  }
}

// ---------------------------------------------------------------------------
// GEMM:  out[r,o] = sum_c A[r,c] * Wt[o,c] + bias[o]
//   A: [4096,1024] bf16 row-major.  Wt: [1024,1024] bf16 (rows = B columns).
//   out_mode 0/1: bf16 [b,h,w,d]   (q / k, attention-friendly)
//   out_mode 2  : bf16 [b,h,d,w]   (V transposed, B-frag friendly for P@V)
//   out_mode 3  : f32  [b,w,c]     (final output)
//   WG = 256 thr = 8 waves, 4x2 wave grid; wave tile 32x64 -> WG tile 128x128.
// ---------------------------------------------------------------------------
__global__ __launch_bounds__(256)
void gemm_bf16(const __bf16* __restrict__ A, const __bf16* __restrict__ Wt,
               const float* __restrict__ bias, __bf16* __restrict__ out_bf,
               float* __restrict__ out_f, int out_mode) {
  int lane = threadIdx.x & 31;
  int wid  = threadIdx.x >> 5;
  int row0 = blockIdx.x * 128 + (wid & 3) * 32;
  int col0 = blockIdx.y * 128 + (wid >> 2) * 64;
  int hi = lane >> 4, nr = lane & 15;

  v8f acc[2][4] = {};

  for (int kk = 0; kk < BINS; kk += 32) {
    // speculative prefetch of the next K-slab (-> global_prefetch_b8)
    __builtin_prefetch(A + (size_t)(row0 + nr) * BINS + kk + 64, 0, 1);
    __builtin_prefetch(Wt + (size_t)(col0 + nr) * BINS + kk + 64, 0, 1);

    v16bf a0 = load_a_frag(A, BINS, row0,      kk, lane);
    v16bf a1 = load_a_frag(A, BINS, row0 + 16, kk, lane);
    v16bf b0 = load_b_frag(Wt, BINS, kk, col0,      lane);
    v16bf b1 = load_b_frag(Wt, BINS, kk, col0 + 16, lane);
    v16bf b2 = load_b_frag(Wt, BINS, kk, col0 + 32, lane);
    v16bf b3 = load_b_frag(Wt, BINS, kk, col0 + 48, lane);

    acc[0][0] = WMMA_BF16(a0, b0, acc[0][0]);
    acc[0][1] = WMMA_BF16(a0, b1, acc[0][1]);
    acc[0][2] = WMMA_BF16(a0, b2, acc[0][2]);
    acc[0][3] = WMMA_BF16(a0, b3, acc[0][3]);
    acc[1][0] = WMMA_BF16(a1, b0, acc[1][0]);
    acc[1][1] = WMMA_BF16(a1, b1, acc[1][1]);
    acc[1][2] = WMMA_BF16(a1, b2, acc[1][2]);
    acc[1][3] = WMMA_BF16(a1, b3, acc[1][3]);
  }

  // Epilogue. C layout: VGPR v -> (M = v + 8*hi, N = lane&15).
  #pragma unroll
  for (int i = 0; i < 2; i++) {
    #pragma unroll
    for (int n = 0; n < 4; n++) {
      int col = col0 + n * 16 + nr;
      float bv = bias[col];
      #pragma unroll
      for (int v = 0; v < 8; v++) {
        int row = row0 + i * 16 + v + hi * 8;
        float val = acc[i][n][v] + bv;
        if (out_mode == 3) {
          out_f[(size_t)row * BINS + col] = val;
        } else {
          int bidx = row >> 10, wloc = row & 1023;
          int hh = col >> 6,  dd = col & 63;
          size_t addr;
          if (out_mode == 2)
            addr = (((size_t)bidx * HEADS + hh) * HEADDIM + dd) * CROP + wloc;
          else
            addr = (((size_t)bidx * HEADS + hh) * CROP + wloc) * HEADDIM + dd;
          out_bf[addr] = to_bf16(val);
        }
      }
    }
  }
}

// ---------------------------------------------------------------------------
// Flash attention per (b,h).  WG = 8 waves, each wave owns 16 query rows
// (128 queries per WG).  Key loop in blocks of 32.
//   S tile (16x32) = q@k^T + erT@q^T   (8 WMMAs, accumulated in-register)
//   online softmax (width-16 shuffles; each C-layout row lives in one half)
//   P (bf16, A-layout via per-wave LDS bounce) @ V^T-frag  (4 WMMAs)
// ---------------------------------------------------------------------------
__global__ __launch_bounds__(256)
void mb_attention(const __bf16* __restrict__ qg, const __bf16* __restrict__ kg,
                  const __bf16* __restrict__ vtg, const __bf16* __restrict__ erT,
                  __bf16* __restrict__ av) {
  // per-wave P staging: 16 rows x 32 keys bf16, padded row stride 40 (=80B,
  // 16B aligned) to keep ds_load_b128 aligned and stagger banks.
  __shared__ __align__(16) __bf16 ldsP[8][16][40];

  int lane = threadIdx.x & 31;
  int wid  = threadIdx.x >> 5;
  int hi = lane >> 4, lr = lane & 15;
  int bh = blockIdx.y;                      // 0..63 = b*16+h
  int q0 = blockIdx.x * 128 + wid * 16;     // first query row of this wave

  const __bf16* qb = qg  + (size_t)bh * CROP * HEADDIM;   // [w][64]
  const __bf16* kb = kg  + (size_t)bh * CROP * HEADDIM;   // [w][64]
  const __bf16* vt = vtg + (size_t)bh * HEADDIM * CROP;   // [64][w]

  // Q and erT A-frags for this wave's 16 query rows (K = d = 64 -> 2 frags)
  v16bf qa0 = load_a_frag(qb,  HEADDIM, q0, 0,  lane);
  v16bf qa1 = load_a_frag(qb,  HEADDIM, q0, 32, lane);
  v16bf ea0 = load_a_frag(erT, HEADDIM, q0, 0,  lane);
  v16bf ea1 = load_a_frag(erT, HEADDIM, q0, 32, lane);

  float m_r[8], l_r[8];
  #pragma unroll
  for (int v = 0; v < 8; v++) { m_r[v] = -1e30f; l_r[v] = 0.0f; }
  v8f oacc[4] = {};

  const float scale = 0.03125f;   // 1/sqrt(c) = 1/32

  for (int j0 = 0; j0 < CROP; j0 += 32) {
    v8f s[2] = {};
    #pragma unroll
    for (int t = 0; t < 2; t++) {
      int jc = j0 + t * 16;
      // content term: q_i . k_j
      v16bf kb0 = load_b_frag(kb, HEADDIM, 0,  jc, lane);
      v16bf kb1 = load_b_frag(kb, HEADDIM, 32, jc, lane);
      s[t] = WMMA_BF16(qa0, kb0, s[t]);
      s[t] = WMMA_BF16(qa1, kb1, s[t]);
      // positional term: erT_i . q_j   (same accumulator)
      v16bf qB0 = load_b_frag(qb, HEADDIM, 0,  jc, lane);
      v16bf qB1 = load_b_frag(qb, HEADDIM, 32, jc, lane);
      s[t] = WMMA_BF16(ea0, qB0, s[t]);
      s[t] = WMMA_BF16(ea1, qB1, s[t]);
    }

    // online softmax update; row (v + 8*hi) spans 16 lanes of this half
    #pragma unroll
    for (int v = 0; v < 8; v++) {
      float x0 = s[0][v] * scale;
      float x1 = s[1][v] * scale;
      float mx = fmaxf(x0, x1);
      #pragma unroll
      for (int off = 8; off; off >>= 1) mx = fmaxf(mx, __shfl_xor(mx, off, 16));
      float mnew  = fmaxf(m_r[v], mx);
      float alpha = __expf(m_r[v] - mnew);
      float p0 = __expf(x0 - mnew);
      float p1 = __expf(x1 - mnew);
      float rs = p0 + p1;
      #pragma unroll
      for (int off = 8; off; off >>= 1) rs += __shfl_xor(rs, off, 16);
      l_r[v] = l_r[v] * alpha + rs;
      m_r[v] = mnew;
      #pragma unroll
      for (int n = 0; n < 4; n++) oacc[n][v] *= alpha;
      int prow = v + hi * 8;
      ldsP[wid][prow][lr]      = to_bf16(p0);
      ldsP[wid][prow][16 + lr] = to_bf16(p1);
    }

    // Re-read P in A-layout (LDS ops are in-order within a wave).
    {
      int kbase = hi ? 8 : 0;
      union { v16bf bf; v4u u[2]; } u;
      u.u[0] = *(const v4u*)(&ldsP[wid][lr][kbase]);
      u.u[1] = *(const v4u*)(&ldsP[wid][lr][kbase + 16]);
      v16bf pa = u.bf;
      #pragma unroll
      for (int n = 0; n < 4; n++) {
        v16bf vb = load_b_frag(vt, CROP, j0, n * 16, lane);  // B col = d row of V^T
        oacc[n] = WMMA_BF16(pa, vb, oacc[n]);
      }
    }
  }

  // normalize and write av as bf16 [b, w, h*64+d] for the output projection
  int b = bh >> 4, h = bh & 15;
  #pragma unroll
  for (int n = 0; n < 4; n++) {
    #pragma unroll
    for (int v = 0; v < 8; v++) {
      int wrow = q0 + v + hi * 8;
      int d    = n * 16 + lr;
      float val = oacc[n][v] / l_r[v];
      av[((size_t)b * CROP + wrow) * BINS + h * HEADDIM + d] = to_bf16(val);
    }
  }
}

// ---------------------------------------------------------------------------
// host launch
// ---------------------------------------------------------------------------
extern "C" void kernel_launch(void* const* d_in, const int* in_sizes, int n_in,
                              void* d_out, int out_size, void* d_ws, size_t ws_size,
                              hipStream_t stream) {
  (void)in_sizes; (void)n_in; (void)out_size; (void)ws_size;

  const float* x  = (const float*)d_in[0];
  const float* Wq = (const float*)d_in[1];
  const float* bq = (const float*)d_in[2];
  const float* Wk = (const float*)d_in[3];
  const float* bk = (const float*)d_in[4];
  const float* Wv = (const float*)d_in[5];
  const float* bv = (const float*)d_in[6];
  const float* Wo = (const float*)d_in[7];
  const float* bo = (const float*)d_in[8];
  const float* er = (const float*)d_in[9];

  char* ws = (char*)d_ws;
  __bf16* xb  = (__bf16*)(ws);                        // 8 MB  x bf16
  __bf16* wqb = (__bf16*)(ws + ( 8ull << 20));        // 2 MB
  __bf16* wkb = (__bf16*)(ws + (10ull << 20));        // 2 MB
  __bf16* wvb = (__bf16*)(ws + (12ull << 20));        // 2 MB
  __bf16* wob = (__bf16*)(ws + (14ull << 20));        // 2 MB
  __bf16* ert = (__bf16*)(ws + (16ull << 20));        // 128 KB erT [1024][64]
  __bf16* qbf = (__bf16*)(ws + (17ull << 20));        // 8 MB  q  [b,h,w,d]
  __bf16* kbf = (__bf16*)(ws + (25ull << 20));        // 8 MB  k  [b,h,w,d]
  __bf16* vtb = (__bf16*)(ws + (33ull << 20));        // 8 MB  vT [b,h,d,w]
  __bf16* avb = (__bf16*)(ws + (41ull << 20));        // 8 MB  av [b,w,c]
  float* out = (float*)d_out;

  cvt_bf16<<<ROWS * BINS / 256, 256, 0, stream>>>(x,  xb,  ROWS * BINS);
  cvt_bf16<<<BINS * BINS / 256, 256, 0, stream>>>(Wq, wqb, BINS * BINS);
  cvt_bf16<<<BINS * BINS / 256, 256, 0, stream>>>(Wk, wkb, BINS * BINS);
  cvt_bf16<<<BINS * BINS / 256, 256, 0, stream>>>(Wv, wvb, BINS * BINS);
  cvt_bf16<<<BINS * BINS / 256, 256, 0, stream>>>(Wo, wob, BINS * BINS);
  cvt_erT<<<HEADDIM * CROP / 256, 256, 0, stream>>>(er, ert);

  dim3 gg(ROWS / 128, BINS / 128);
  gemm_bf16<<<gg, 256, 0, stream>>>(xb, wqb, bq, qbf, nullptr, 0);   // Q
  gemm_bf16<<<gg, 256, 0, stream>>>(xb, wkb, bk, kbf, nullptr, 1);   // K
  gemm_bf16<<<gg, 256, 0, stream>>>(xb, wvb, bv, vtb, nullptr, 2);   // V^T

  mb_attention<<<dim3(CROP / 128, BATCH * HEADS), 256, 0, stream>>>(qbf, kbf, vtb, ert, avb);

  gemm_bf16<<<gg, 256, 0, stream>>>(avb, wob, bo, nullptr, out, 3);  // out proj
}
